// Decoder_38199439130847
// MI455X (gfx1250) — matrix operand order repaired and checked by reading, not compiled, not run on hardware
//
#include <hip/hip_runtime.h>
#include <hip/hip_bf16.h>
#include <hip/hip_fp16.h>

typedef __attribute__((ext_vector_type(16))) _Float16 v16h;
typedef __attribute__((ext_vector_type(8)))  float    v8f;

#define N_MOL 256
#define N_PER 32
#define NNODE (N_MOL*N_PER)
#define EPM   (N_PER*(N_PER-1))   /* 992 edges per molecule */
#define NEDGE (N_MOL*EPM)
#define SDIM 256
#define VDIM 64
#define EDIM 32
#define NAF 16
#define NBT 5
#define LAT 64
#define LAYERS 5
#define MSG_IN 546
#define MSG_OUT 353
#define BOND_IN 289
#define BOND_HID 128

/* padded-K strides of the pre-converted f16 weight images */
#define K1PAD 576   /* MSG_IN 546 -> 576 */
#define K2PAD 256
#define KBPAD 320   /* BOND_IN 289 -> 320 */
#define W2COLS 320  /* packed cols: 0..255 = m_s, 256..319 -> gc 320..383 (m_p/m_e/pad) */

/* ---- dynamic LDS layout (bytes) ---- */
#define OFF_S    0        /* float    s[32][256]      32768 */
#define OFF_SACC 32768    /* float    s_acc[32][256]  32768 (reused as t1 in head) */
#define OFF_SH   65536    /* _Float16 sh[32][256]     16384 */
#define OFF_HDN  81920    /* _Float16 hdn[64][256]    32768 (reused as float h2[64][128]) */
#define OFF_BST  114688   /* _Float16 bstT[320][32]   20480 (B staged transposed) */
#define OFF_AST  135168   /* _Float16 ast[64][32]      4096 */
#define OFF_GEOM 139264   /* float    geom[64][8]      2048 (d,a,rn0..2) */
#define OFF_POS  141312   /* float    pos[32][4]        512 */
#define OFF_PACC 141824   /* float    pos_acc[32][4]    512 */
#define SMEM_BYTES 142336

/* sigmoid denominator >= 1; raw v_rcp_f32 (~1 ulp) is safe — avoids the
   correctly-rounded div-fixup sequence the compiler emits for '/' / __frcp_rn */
__device__ __forceinline__ float silu_f(float x) {
    return x * __builtin_amdgcn_rcpf(1.0f + __expf(-x));
}

/* fully-connected-minus-diagonal edge enumeration (matches _block_fc_edges) */
__device__ __forceinline__ void edge_pair(int le, int& s, int& t) {
    s = le / 31;
    int r = le - s * 31;
    t = r + (r >= s ? 1 : 0);
}

/* A fragment: 16x32 f16, row-contiguous in LDS (ldstride halfs per row).
   ISA 7.12.2: lanes 0-15 -> K 0..7 / 16..23; lanes 16-31 -> K 8..15 / 24..31. */
__device__ __forceinline__ v16h load_a_frag(const _Float16* base, int ldstride,
                                            int row_base, int lane) {
    const _Float16* p = base + (row_base + (lane & 15)) * ldstride + ((lane & 16) ? 8 : 0);
    v16h a;
#pragma unroll
    for (int j = 0; j < 8; ++j) { a[j] = p[j]; a[j + 8] = p[16 + j]; }
    return a;
}

/* B fragment: 32x16 f16 from column-major staged LDS (bstT[col][k], 32 k per col).
   lanes 0-15 -> K 0..7 / 16..23 of col=lane; lanes 16-31 -> K 8..15 / 24..31 of col=lane-16. */
__device__ __forceinline__ v16h load_b_frag(const _Float16* bstT, int col_base, int lane) {
    const _Float16* p = bstT + (col_base + (lane & 15)) * 32 + ((lane & 16) ? 8 : 0);
    v16h b;
#pragma unroll
    for (int j = 0; j < 8; ++j) { b[j] = p[j]; b[j + 8] = p[16 + j]; }
    return b;
}

/* ---------------- init kernels ---------------- */

__global__ __launch_bounds__(256) void init_nodes(
    const float* __restrict__ x, const float* __restrict__ z,
    const float* __restrict__ rot,
    const float* __restrict__ Wam, const float* __restrict__ bam,
    float* __restrict__ s0, float* __restrict__ pos0)
{
    __shared__ float in[NAF + LAT];
    __shared__ float hp[3];
    const int node = blockIdx.x;
    const int tid = threadIdx.x;
    if (tid < NAF)            in[tid] = x[node * NAF + tid];
    else if (tid < NAF + LAT) in[tid] = z[node * LAT + (tid - NAF)];
    __syncthreads();
    for (int c = tid; c < SDIM + 3; c += 256) {
        float acc = bam[c];
        for (int k = 0; k < NAF + LAT; ++k) acc += in[k] * Wam[k * (SDIM + 3) + c];
        if (c < SDIM) s0[(size_t)node * SDIM + c] = acc;
        else          hp[c - SDIM] = acc;
    }
    __syncthreads();
    if (tid < 3) {
        const float* R = rot + (node >> 5) * 9 + tid * 3;
        pos0[node * 3 + tid] = R[0] * hp[0] + R[1] * hp[1] + R[2] * hp[2];
    }
}

__global__ __launch_bounds__(256) void init_edges(
    const float* __restrict__ ea, const float* __restrict__ Wbm,
    const float* __restrict__ bbm, float* __restrict__ eG)
{
    size_t i = (size_t)blockIdx.x * 256 + threadIdx.x;
    if (i >= (size_t)NEDGE * EDIM) return;
    int e = (int)(i >> 5), c = (int)(i & 31);
    float acc = bbm[c];
#pragma unroll
    for (int k = 0; k < NBT; ++k) acc += ea[e * NBT + k] * Wbm[k * EDIM + c];
    eG[i] = acc;
}

/* pre-transpose + f32->f16 convert + K-pad the WMMA weight operands (done once) */

__global__ __launch_bounds__(256) void conv_w1(const float* __restrict__ gW1,
                                               _Float16* __restrict__ w1t)
{
    int idx = blockIdx.x * 256 + threadIdx.x;
    if (idx >= LAYERS * SDIM * K1PAD) return;
    int l = idx / (SDIM * K1PAD);
    int rem = idx - l * (SDIM * K1PAD);
    int c = rem / K1PAD, k = rem - c * K1PAD;
    float v = (k < MSG_IN) ? gW1[(size_t)l * MSG_IN * SDIM + (size_t)k * SDIM + c] : 0.f;
    w1t[idx] = (_Float16)v;
}

__global__ __launch_bounds__(256) void conv_w2(const float* __restrict__ gW2,
                                               _Float16* __restrict__ w2t)
{
    int idx = blockIdx.x * 256 + threadIdx.x;
    if (idx >= LAYERS * W2COLS * K2PAD) return;
    int l = idx / (W2COLS * K2PAD);
    int rem = idx - l * (W2COLS * K2PAD);
    int c2 = rem / K2PAD, k = rem - c2 * K2PAD;
    int gc = (c2 < SDIM) ? c2 : c2 + VDIM;       /* skip dead m_v columns 256..319 */
    float v = (gc < MSG_OUT) ? gW2[(size_t)l * SDIM * MSG_OUT + (size_t)k * MSG_OUT + gc] : 0.f;
    w2t[idx] = (_Float16)v;
}

__global__ __launch_bounds__(256) void conv_we1(const float* __restrict__ We1,
                                                _Float16* __restrict__ we1t)
{
    int idx = blockIdx.x * 256 + threadIdx.x;
    if (idx >= BOND_HID * KBPAD) return;
    int c = idx / KBPAD, k = idx - c * KBPAD;
    float v = (k < BOND_IN) ? We1[(size_t)k * BOND_HID + c] : 0.f;
    we1t[idx] = (_Float16)v;
}

/* ---------------- main persistent per-molecule kernel ---------------- */

__global__ __launch_bounds__(256) void decoder_main(
    const _Float16* __restrict__ w1t, const float* __restrict__ gb1,
    const _Float16* __restrict__ w2t, const float* __restrict__ gb2,
    const float* __restrict__ Wh1, const float* __restrict__ bh1,
    const float* __restrict__ Wh2, const float* __restrict__ bh2,
    const _Float16* __restrict__ we1t, const float* __restrict__ be1,
    const float* __restrict__ We2, const float* __restrict__ be2,
    const float* __restrict__ s0,  const float* __restrict__ pos0,
    float* __restrict__ eG,
    float* __restrict__ atoms_out, float* __restrict__ bonds_out,
    float* __restrict__ pos_out)
{
    extern __shared__ char smem[];
    float*    sS   = (float*)(smem + OFF_S);
    float*    sAcc = (float*)(smem + OFF_SACC);
    _Float16* sH   = (_Float16*)(smem + OFF_SH);
    _Float16* hdn  = (_Float16*)(smem + OFF_HDN);
    _Float16* bstT = (_Float16*)(smem + OFF_BST);
    _Float16* ast  = (_Float16*)(smem + OFF_AST);
    float*    geom = (float*)(smem + OFF_GEOM);
    float*    sPos = (float*)(smem + OFF_POS);
    float*    pAcc = (float*)(smem + OFF_PACC);
    uint4*    bst4 = (uint4*)(smem + OFF_BST);
    uint4*    ast4 = (uint4*)(smem + OFF_AST);

    const int b    = blockIdx.x;
    const int tid  = threadIdx.x;
    const int wave = tid >> 5, lane = tid & 31;
    const int msub = wave & 3;              /* 4 m-subtiles per 64-edge tile */
    const uint4 zero4 = make_uint4(0u, 0u, 0u, 0u);

    /* per-thread fixed (row, 8-half slot) for vectorized A staging: 64 rows x 4 slots */
    const int arow = tid >> 2;              /* edge row 0..63  */
    const int aq   = tid & 3;               /* 8-half slot 0..3 */

    /* load persistent state for this molecule */
    for (int i = tid; i < N_PER * SDIM; i += 256) sS[i] = s0[(size_t)b * N_PER * SDIM + i];
    for (int i = tid; i < N_PER * 4; i += 256) {
        int node = i >> 2, xx = i & 3;
        sPos[i] = (xx < 3) ? pos0[(b * N_PER + node) * 3 + xx] : 0.f;
    }
    __syncthreads();

    for (int layer = 0; layer < LAYERS; ++layer) {
        const _Float16* W1L = w1t + (size_t)layer * SDIM * K1PAD;
        const float*    b1  = gb1 + layer * SDIM;
        const _Float16* W2L = w2t + (size_t)layer * W2COLS * K2PAD;
        const float*    b2  = gb2 + layer * MSG_OUT;

        for (int i = tid; i < N_PER * SDIM; i += 256) { sH[i] = (_Float16)sS[i]; sAcc[i] = 0.f; }
        for (int i = tid; i < N_PER * 4; i += 256) pAcc[i] = 0.f;
        __syncthreads();

        for (int mt = 0; mt < 16; ++mt) {
            const int m0 = mt * 64;

            /* ---- geometry for the 64 edges of this tile ---- */
            if (tid < 64) {
                int le = m0 + tid;
                float d = 0.f, a = 0.f, rn0 = 0.f, rn1 = 0.f, rn2 = 0.f;
                if (le < EPM) {
                    int es, et; edge_pair(le, es, et);
                    float px0 = sPos[es*4+0], px1 = sPos[es*4+1], px2 = sPos[es*4+2];
                    float py0 = sPos[et*4+0], py1 = sPos[et*4+1], py2 = sPos[et*4+2];
                    float r0 = py0 - px0, r1 = py1 - px1, r2 = py2 - px2;
                    a = py0*px0 + py1*px1 + py2*px2;
                    d = sqrtf(fmaxf(r0*r0 + r1*r1 + r2*r2, 1e-6f));
                    float inv = 1.0f / (1.0f + d);
                    rn0 = r0*inv; rn1 = r1*inv; rn2 = r2*inv;
                }
                float* g = geom + tid * 8;
                g[0] = d; g[1] = a; g[2] = rn0; g[3] = rn1; g[4] = rn2;
            }
            __syncthreads();

            /* per-thread edge identity for A staging */
            const int aLe = m0 + arow;
            const bool aValid = (aLe < EPM);
            int aSrc, aTgt; edge_pair(aLe < EPM ? aLe : 0, aSrc, aTgt);

            /* ---- GEMM1: hdn[64,256] = silu(feat[64,546] @ W1 + b1) ---- */
            v8f acc1[8] = {};
            const int n0 = (wave >> 2) * 128;     /* waves 0-3: n 0..127, 4-7: 128..255 */
            for (int kc = 0; kc < 18; ++kc) {     /* K = 546 padded to 576 */
                const int k0 = kc * 32;
                /* stage A (64x32): vectorized b128 copy from sH for kc<16,
                   scalar only for the tiny e / d / a chunks */
                if (kc < 16) {
                    int node = (kc < 8) ? aSrc : aTgt;
                    ast4[tid] = aValid
                        ? *(const uint4*)(sH + node * SDIM + (k0 & 255) + aq * 8)
                        : zero4;
                } else if (kc == 16) {            /* gk 512..543 = e[edge][0..31] */
                    for (int i = tid; i < 64 * 32; i += 256) {
                        int row = i >> 5, k = i & 31;
                        int le = m0 + row;
                        ast[i] = (le < EPM)
                            ? (_Float16)eG[((size_t)b*EPM + le)*EDIM + k]
                            : (_Float16)0.f;
                    }
                } else {                          /* gk 544..575: d, a, zero pad */
                    for (int i = tid; i < 64 * 32; i += 256) {
                        int row = i >> 5, k = i & 31;
                        float v = 0.f;
                        if (k == 0)      v = geom[row*8 + 0];
                        else if (k == 1) v = geom[row*8 + 1];
                        ast[i] = (_Float16)v;
                    }
                }
                /* stage B: straight 128-bit copy from pre-transposed f16 W1 image */
                for (int i = tid; i < (SDIM * 32) / 8; i += 256) {     /* 1024 uint4 */
                    int col = i >> 2, q = i & 3;
                    bst4[i] = *(const uint4*)(W1L + col * K1PAD + k0 + q * 8);
                }
                __syncthreads();
                v16h af = load_a_frag(ast, 32, msub * 16, lane);
#pragma unroll
                for (int t = 0; t < 8; ++t) {
                    v16h bf = load_b_frag(bstT, n0 + t * 16, lane);
                    acc1[t] = __builtin_amdgcn_wmma_f32_16x16x32_f16(
                        false, af, false, bf, (short)0, acc1[t], false, false);
                }
                __syncthreads();
            }
            /* epilogue 1: bias + silu -> f16 hdn */
            {
                int r8 = (lane & 16) ? 8 : 0;
#pragma unroll
                for (int t = 0; t < 8; ++t) {
                    int c = n0 + t * 16 + (lane & 15);
                    float bias = b1[c];
#pragma unroll
                    for (int j = 0; j < 8; ++j) {
                        int row = msub * 16 + r8 + j;
                        hdn[row * SDIM + c] = (_Float16)silu_f(acc1[t][j] + bias);
                    }
                }
            }
            __syncthreads();

            /* ---- GEMM2: m = hdn @ W2 + b2 over packed live columns ---- */
            v8f acc2[10] = {};
            const int ntBeg = (wave < 4) ? 0 : 10;   /* 20 n-subtiles, 10 per wave group */
            for (int kc = 0; kc < 8; ++kc) {         /* K = 256 */
                const int k0 = kc * 32;
                for (int i = tid; i < (W2COLS * 32) / 8; i += 256) {   /* 1280 uint4 */
                    int col = i >> 2, q = i & 3;
                    bst4[i] = *(const uint4*)(W2L + col * K2PAD + k0 + q * 8);
                }
                __syncthreads();
                v16h af = load_a_frag(hdn + k0, SDIM, msub * 16, lane);
#pragma unroll
                for (int q = 0; q < 10; ++q) {
                    v16h bf = load_b_frag(bstT, (ntBeg + q) * 16, lane);
                    acc2[q] = __builtin_amdgcn_wmma_f32_16x16x32_f16(
                        false, af, false, bf, (short)0, acc2[q], false, false);
                }
                __syncthreads();
            }
            /* epilogue 2: segment-sum scatter (LDS atomics) + e update (exclusive RMW) */
            {
                int r8 = (lane & 16) ? 8 : 0;
#pragma unroll
                for (int q = 0; q < 10; ++q) {
                    int c2 = (ntBeg + q) * 16 + (lane & 15);
                    int gc = (c2 < SDIM) ? c2 : c2 + VDIM;
                    if (gc < MSG_OUT) {
                        float bias = b2[gc];
#pragma unroll
                        for (int j = 0; j < 8; ++j) {
                            int rloc = msub * 16 + r8 + j;
                            int me = m0 + rloc;
                            if (me < EPM) {
                                float val = acc2[q][j] + bias;
                                int es, et; edge_pair(me, es, et);
                                if (gc < SDIM) {
                                    atomicAdd(&sAcc[et * SDIM + gc], val);
                                } else if (gc == SDIM + VDIM) {        /* m_p */
                                    const float* g = geom + rloc * 8;
                                    atomicAdd(&pAcc[et*4+0], g[2] * val);
                                    atomicAdd(&pAcc[et*4+1], g[3] * val);
                                    atomicAdd(&pAcc[et*4+2], g[4] * val);
                                } else {                                /* m_e 321..352 */
                                    size_t ei = ((size_t)b*EPM + me)*EDIM + (gc - (SDIM+VDIM+1));
                                    eG[ei] += val;
                                }
                            }
                        }
                    }
                }
            }
            __syncthreads();
        } /* mt */

        /* layer state update: seg_mean divisor is uniformly 31 */
        const float invc = 1.0f / 31.0f;
        for (int i = tid; i < N_PER * SDIM; i += 256) sS[i] += sAcc[i] * invc;
        for (int i = tid; i < N_PER * 4; i += 256) if ((i & 3) < 3) sPos[i] += pAcc[i] * invc;
        __threadfence();          /* make eG writes visible to next layer's reads */
        __syncthreads();
    } /* layer */

    /* ---- atom head: atoms = silu(s@Wh1+bh1)@Wh2+bh2 (VALU, tiny) ---- */
    float* t1 = sAcc;
    for (int i = tid; i < N_PER * SDIM; i += 256) {
        int node = i >> 8, c = i & 255;
        float acc = bh1[c];
        for (int k = 0; k < SDIM; ++k) acc += sS[node * SDIM + k] * Wh1[(size_t)k * SDIM + c];
        t1[i] = silu_f(acc);
    }
    __syncthreads();
    for (int i = tid; i < N_PER * NAF; i += 256) {
        int node = i >> 4, c = i & 15;
        float acc = bh2[c];
        for (int k = 0; k < SDIM; ++k) acc += t1[node * SDIM + k] * Wh2[k * NAF + c];
        atoms_out[((size_t)b * N_PER + node) * NAF + c] = acc;
    }
    for (int i = tid; i < N_PER * 3; i += 256) {
        int node = i / 3, xx = i - node * 3;
        pos_out[((size_t)b * N_PER + node) * 3 + xx] = sPos[node * 4 + xx];
    }
    __syncthreads();

    /* ---- bond head: bonds = silu(ef@We1+be1)@We2+be2 ---- */
    float* h2 = (float*)(smem + OFF_HDN);
    for (int mt = 0; mt < 16; ++mt) {
        const int m0 = mt * 64;
        if (tid < 64) {
            int le = m0 + tid;
            float d = 0.f;
            if (le < EPM) {
                int es, et; edge_pair(le, es, et);
                float r0 = sPos[et*4+0] - sPos[es*4+0];
                float r1 = sPos[et*4+1] - sPos[es*4+1];
                float r2 = sPos[et*4+2] - sPos[es*4+2];
                d = sqrtf(fmaxf(r0*r0 + r1*r1 + r2*r2, 1e-6f));
            }
            geom[tid * 8] = d;
        }
        __syncthreads();

        const int aLe = m0 + arow;
        const bool aValid = (aLe < EPM);
        int aSrc, aTgt; edge_pair(aLe < EPM ? aLe : 0, aSrc, aTgt);

        v8f accb[4] = {};
        const int nb0 = (wave >> 2) * 64;   /* N=128: waves 0-3 cols 0..63, 4-7 cols 64..127 */
        for (int kc = 0; kc < 10; ++kc) {   /* K = 289 padded to 320 */
            const int k0 = kc * 32;
            if (kc < 8) {                    /* s[src]+s[tgt]: 8-half sum-gather per thread */
                const float* ps = sS + aSrc * SDIM + k0 + aq * 8;
                const float* pt = sS + aTgt * SDIM + k0 + aq * 8;
                _Float16* dst = ast + arow * 32 + aq * 8;
#pragma unroll
                for (int j = 0; j < 8; ++j)
                    dst[j] = aValid ? (_Float16)(ps[j] + pt[j]) : (_Float16)0.f;
            } else if (kc == 8) {            /* gk 256..287 = e[edge][0..31] */
                for (int i = tid; i < 64 * 32; i += 256) {
                    int row = i >> 5, k = i & 31;
                    int le = m0 + row;
                    ast[i] = (le < EPM)
                        ? (_Float16)eG[((size_t)b*EPM + le)*EDIM + k]
                        : (_Float16)0.f;
                }
            } else {                         /* gk 288..319: d, zero pad */
                for (int i = tid; i < 64 * 32; i += 256) {
                    int row = i >> 5, k = i & 31;
                    ast[i] = (_Float16)((k == 0) ? geom[row*8] : 0.f);
                }
            }
            for (int i = tid; i < (BOND_HID * 32) / 8; i += 256) {     /* 512 uint4 */
                int col = i >> 2, q = i & 3;
                bst4[i] = *(const uint4*)(we1t + col * KBPAD + k0 + q * 8);
            }
            __syncthreads();
            v16h af = load_a_frag(ast, 32, msub * 16, lane);
#pragma unroll
            for (int t = 0; t < 4; ++t) {
                v16h bf = load_b_frag(bstT, nb0 + t * 16, lane);
                accb[t] = __builtin_amdgcn_wmma_f32_16x16x32_f16(
                    false, af, false, bf, (short)0, accb[t], false, false);
            }
            __syncthreads();
        }
        {
            int r8 = (lane & 16) ? 8 : 0;
#pragma unroll
            for (int t = 0; t < 4; ++t) {
                int c = nb0 + t * 16 + (lane & 15);
                float bias = be1[c];
#pragma unroll
                for (int j = 0; j < 8; ++j) {
                    int row = msub * 16 + r8 + j;
                    h2[row * BOND_HID + c] = silu_f(accb[t][j] + bias);
                }
            }
        }
        __syncthreads();
        for (int i = tid; i < 64 * NBT; i += 256) {
            int r = i / NBT, c = i - r * NBT;
            int le = m0 + r;
            if (le < EPM) {
                float acc = be2[c];
                for (int k = 0; k < BOND_HID; ++k) acc += h2[r * BOND_HID + k] * We2[k * NBT + c];
                bonds_out[((size_t)b * EPM + le) * NBT + c] = acc;
            }
        }
        __syncthreads();
    }
}

/* ---------------- host entry ---------------- */

extern "C" void kernel_launch(void* const* d_in, const int* in_sizes, int n_in,
                              void* d_out, int out_size, void* d_ws, size_t ws_size,
                              hipStream_t stream)
{
    (void)in_sizes; (void)n_in; (void)out_size; (void)ws_size;
    const float* x   = (const float*)d_in[0];
    const float* z   = (const float*)d_in[1];
    const float* rot = (const float*)d_in[2];
    /* d_in[3] edge_index, d_in[5] batch: structure is analytic, unused */
    const float* ea  = (const float*)d_in[4];
    const float* Wam = (const float*)d_in[6];
    const float* bam = (const float*)d_in[7];
    const float* Wbm = (const float*)d_in[8];
    const float* bbm = (const float*)d_in[9];
    const float* gW1 = (const float*)d_in[10];
    const float* gb1 = (const float*)d_in[11];
    const float* gW2 = (const float*)d_in[12];
    const float* gb2 = (const float*)d_in[13];
    const float* Wh1 = (const float*)d_in[14];
    const float* bh1 = (const float*)d_in[15];
    const float* Wh2 = (const float*)d_in[16];
    const float* bh2 = (const float*)d_in[17];
    const float* We1 = (const float*)d_in[18];
    const float* be1 = (const float*)d_in[19];
    const float* We2 = (const float*)d_in[20];
    const float* be2 = (const float*)d_in[21];

    float* s0   = (float*)d_ws;                       /* 8192*256 f32 */
    float* pos0 = s0 + (size_t)NNODE * SDIM;          /* 8192*3  f32 */
    float* eG   = pos0 + (size_t)NNODE * 3;           /* E*32    f32 (L2-resident) */
    _Float16* w1t  = (_Float16*)(eG + (size_t)NEDGE * EDIM);       /* [5][256][576] */
    _Float16* w2t  = w1t + (size_t)LAYERS * SDIM * K1PAD;          /* [5][320][256] */
    _Float16* we1t = w2t + (size_t)LAYERS * W2COLS * K2PAD;        /* [128][320]    */

    float* atoms = (float*)d_out;
    float* bonds = atoms + (size_t)NNODE * NAF;
    float* posO  = bonds + (size_t)NEDGE * NBT;

    init_nodes<<<NNODE, 256, 0, stream>>>(x, z, rot, Wam, bam, s0, pos0);
    init_edges<<<(int)(((size_t)NEDGE * EDIM + 255) / 256), 256, 0, stream>>>(ea, Wbm, bbm, eG);
    conv_w1 <<<(LAYERS * SDIM * K1PAD + 255) / 256, 256, 0, stream>>>(gW1, w1t);
    conv_w2 <<<(LAYERS * W2COLS * K2PAD + 255) / 256, 256, 0, stream>>>(gW2, w2t);
    conv_we1<<<(BOND_HID * KBPAD + 255) / 256, 256, 0, stream>>>(We1, we1t);

    (void)hipFuncSetAttribute((const void*)decoder_main,
                              hipFuncAttributeMaxDynamicSharedMemorySize, SMEM_BYTES);
    decoder_main<<<N_MOL, 256, SMEM_BYTES, stream>>>(
        w1t, gb1, w2t, gb2, Wh1, bh1, Wh2, bh2,
        we1t, be1, We2, be2, s0, pos0, eG,
        atoms, bonds, posO);
}
